// ALIGNN_72894184948210
// MI455X (gfx1250) — matrix operand order repaired and checked by reading, not compiled, not run on hardware
//
#include <hip/hip_runtime.h>

#define DIM 128
#define EPSV 1e-5f

typedef __attribute__((ext_vector_type(16))) __bf16 bf16x16;
typedef __attribute__((ext_vector_type(8)))  float  f32x8;
typedef __attribute__((ext_vector_type(4)))  unsigned int u32x4;
typedef __attribute__((ext_vector_type(4)))  int i32x4;
typedef __attribute__((ext_vector_type(8)))  int i32x8;

union bf16x16_u { bf16x16 v; unsigned short s[16]; uint4 q[2]; };

#if defined(__has_builtin)
#if __has_builtin(__builtin_amdgcn_tensor_load_to_lds) && __has_builtin(__builtin_amdgcn_s_wait_tensorcnt)
#define USE_TDM 1
#endif
#endif
#ifndef USE_TDM
#define USE_TDM 0
#endif

__device__ __forceinline__ unsigned short f2bf_bits(float f) {
    union { float f; unsigned u; } v; v.f = f;
    return (unsigned short)((v.u + 0x7FFFu + ((v.u >> 16) & 1u)) >> 16);
}
__device__ __forceinline__ float bf2f(unsigned short s) {
    union { unsigned u; float f; } v; v.u = ((unsigned)s) << 16;
    return v.f;
}

// ---- WMMA fragment loaders (wave32 16x16x32 bf16 layouts) ----

// A: 16x32 tile slice; lane row m=lane&15, K-half kh=lane>>4.
// halves 0..7 -> K=kbase+kh*8+j ; halves 8..15 -> K=kbase+16+kh*8+j
// Feature rows are stored as bf16 bits -> two raw 16B loads, no conversion.
__device__ __forceinline__ bf16x16 load_a_frag_bf(const unsigned short* row, int kbase, int kh) {
    bf16x16_u u;
    u.q[0] = *(const uint4*)(row + kbase + kh * 8);
    u.q[1] = *(const uint4*)(row + kbase + 16 + kh * 8);
    return u.v;
}

// A from an f32 row (readout only): convert on the fly.
__device__ __forceinline__ bf16x16 load_a_frag_f32(const float* row, int kbase, int kh) {
    const float4* p0 = (const float4*)(row + kbase + kh * 8);
    const float4* p1 = (const float4*)(row + kbase + 16 + kh * 8);
    float4 x0 = p0[0], x1 = p0[1], y0 = p1[0], y1 = p1[1];
    bf16x16_u u;
    u.s[0] = f2bf_bits(x0.x);  u.s[1] = f2bf_bits(x0.y);
    u.s[2] = f2bf_bits(x0.z);  u.s[3] = f2bf_bits(x0.w);
    u.s[4] = f2bf_bits(x1.x);  u.s[5] = f2bf_bits(x1.y);
    u.s[6] = f2bf_bits(x1.z);  u.s[7] = f2bf_bits(x1.w);
    u.s[8] = f2bf_bits(y0.x);  u.s[9] = f2bf_bits(y0.y);
    u.s[10] = f2bf_bits(y0.z); u.s[11] = f2bf_bits(y0.w);
    u.s[12] = f2bf_bits(y1.x); u.s[13] = f2bf_bits(y1.y);
    u.s[14] = f2bf_bits(y1.z); u.s[15] = f2bf_bits(y1.w);
    return u.v;
}

// B: 32x16 tile from LDS weights stored transposed [n][k]; lane n=lane&15, kg=lane>>4:
// halves 0..7 -> K=kbase+kg*8+j ; halves 8..15 -> K=kbase+16+kg*8+j (matches A).
__device__ __forceinline__ bf16x16 load_b_frag(const unsigned short* WtLds, int n0, int kbase, int kg) {
    const unsigned short* p = WtLds + n0 * DIM + kbase + kg * 8;
    bf16x16_u u;
    u.q[0] = *(const uint4*)(p);
    u.q[1] = *(const uint4*)(p + 16);
    return u.v;
}

__device__ __forceinline__ f32x8 wmma_bf16(bf16x16 a, bf16x16 b, f32x8 c) {
    return __builtin_amdgcn_wmma_f32_16x16x32_bf16(false, a, false, b, (short)0, c, false, false);
}

// ---- TDM: stage `halves` contiguous bf16 elements global -> LDS (1-row tensor) ----
#if USE_TDM
__device__ __forceinline__ void tdm_load_to_lds(const void* gptr, unsigned lds_addr, unsigned halves) {
    unsigned long long ga = (unsigned long long)gptr;
    u32x4 g0;
    g0[0] = 1u;                                               // count=1, user desc
    g0[1] = lds_addr;                                         // LDS byte address
    g0[2] = (unsigned)(ga & 0xFFFFFFFFu);                     // global_addr[31:0]
    g0[3] = (unsigned)((ga >> 32) & 0x01FFFFFFu) | 0x80000000u; // addr[56:32] | type=2
    i32x8 g1;
    g1[0] = (int)0x00010000u;                                 // data_size=1 (2B)
    g1[1] = (int)((halves & 0xFFFFu) << 16);                  // tensor_dim0[15:0]
    g1[2] = (int)(((halves >> 16) & 0xFFFFu) | (1u << 16));   // dim0[31:16] | tensor_dim1=1
    g1[3] = (int)((halves & 0xFFFFu) << 16);                  // tile_dim0 (<=65535)
    g1[4] = 1;                                                // tile_dim1=1
    g1[5] = (int)halves;                                      // tensor_dim0_stride[31:0]
    g1[6] = (int)((halves & 0xFFFFu) << 16);                  // stride0[47:32]=0 | stride1[15:0]
    g1[7] = (int)((halves >> 16) & 0xFFFFu);                  // stride1 high
    i32x4 z4 = {0, 0, 0, 0};
#if __clang_major__ >= 23
    i32x8 z8 = {0, 0, 0, 0, 0, 0, 0, 0};
    __builtin_amdgcn_tensor_load_to_lds(g0, g1, z4, z4, z8, 0);
#else
    __builtin_amdgcn_tensor_load_to_lds(g0, g1, z4, z4, 0);
#endif
}
#endif

__device__ __forceinline__ void stage_weights(const unsigned short* W, char* smem, unsigned halves) {
#if USE_TDM
    if (threadIdx.x < 32) {                                   // wave 0 drives the TDM
        tdm_load_to_lds(W, (unsigned)(size_t)(void*)smem, halves);
        __builtin_amdgcn_s_wait_tensorcnt(0);
    }
    __syncthreads();
#else
    uint4* dst = (uint4*)smem;
    const uint4* src = (const uint4*)W;
    unsigned n16 = halves >> 3;
    for (unsigned i = threadIdx.x; i < n16; i += blockDim.x) dst[i] = src[i];
    __syncthreads();
#endif
}

// ---------------- prep / basis kernels ----------------

__global__ void prep_weights_kernel(const float* __restrict__ W, unsigned short* __restrict__ Wt, int nmats) {
    long idx = (long)blockIdx.x * blockDim.x + threadIdx.x;
    long total = (long)nmats * DIM * DIM;
    if (idx >= total) return;
    long mat = idx / (DIM * DIM);
    int rem = (int)(idx - mat * DIM * DIM);
    int n = rem >> 7, k = rem & 127;
    Wt[idx] = f2bf_bits(W[mat * DIM * DIM + (long)k * DIM + n]);
}

__global__ void embed_kernel(const int* __restrict__ species, const float* __restrict__ table,
                             unsigned short* __restrict__ h, int N) {
    long idx = (long)blockIdx.x * blockDim.x + threadIdx.x;
    if (idx >= (long)N * DIM) return;
    int n = (int)(idx >> 7), d = (int)(idx & 127);
    h[idx] = f2bf_bits(table[species[n] * DIM + d]);
}

__global__ void bessel_kernel(const float* __restrict__ xb, unsigned short* __restrict__ h, int E) {
    long idx = (long)blockIdx.x * blockDim.x + threadIdx.x;
    if (idx >= (long)E * DIM) return;
    long e = idx >> 7; int d = (int)(idx & 127);
    float xs = xb[e] + EPSV;
    const float coef = 0.8164965809277260f;   // sqrt(2/3)
    const float pi_c = 1.0471975511965976f;   // pi/3
    h[idx] = f2bf_bits(coef * __sinf((float)(d + 1) * pi_c * xs) / xs);
}

__global__ void gaussian_kernel(const float* __restrict__ xa, unsigned short* __restrict__ h, int E) {
    long idx = (long)blockIdx.x * blockDim.x + threadIdx.x;
    if (idx >= (long)E * DIM) return;
    long e = idx >> 7; int d = (int)(idx & 127);
    float cx = __cosf(xa[e]);
    float mu = -1.0f + (2.0f / 127.0f) * (float)d;
    float diff = (cx - mu) * 63.5f;           // /step, step = 2/127
    h[idx] = f2bf_bits(__expf(-0.5f * diff * diff));
}

// ---------------- node GEMM: mode 0 -> out = x@W+b ; mode 1 -> x += silu(x@W+b + num/(den+eps))

__global__ void node_mm_kernel(const unsigned short* x, const unsigned short* __restrict__ Wt,
                               const float* __restrict__ bias, unsigned short* out,
                               const float* __restrict__ num, const float* __restrict__ den,
                               int N, int mode) {
    extern __shared__ char smem[];
    unsigned short* sW = (unsigned short*)smem;  // 128x128 bf16 (32 KB)
    stage_weights(Wt, smem, DIM * DIM);
    int wave = threadIdx.x >> 5, lane = threadIdx.x & 31;
    long tileBase = ((long)blockIdx.x * 8 + wave) * 16;
    if (tileBase >= N) return;
    int m = lane & 15, kh = lane >> 4;
    long rowA = tileBase + m; if (rowA > N - 1) rowA = N - 1;
    const unsigned short* xr = x + rowA * DIM;
    bf16x16 A[4];
    #pragma unroll
    for (int ks = 0; ks < 4; ++ks) A[ks] = load_a_frag_bf(xr, ks * 32, kh);
    int n_ = lane & 15, rh = lane >> 4;
    for (int ct = 0; ct < 8; ++ct) {
        int col = ct * 16 + n_;
        float bv = bias[col];
        f32x8 acc;
        #pragma unroll
        for (int r = 0; r < 8; ++r) acc[r] = bv;
        #pragma unroll
        for (int ks = 0; ks < 4; ++ks)
            acc = wmma_bf16(A[ks], load_b_frag(sW, col, ks * 32, rh), acc);
        #pragma unroll
        for (int r = 0; r < 8; ++r) {
            long row = tileBase + rh * 8 + r;
            if (row < N) {
                long o = row * DIM + col;
                float y = acc[r];
                if (mode == 0) {
                    out[o] = f2bf_bits(y);
                } else {
                    float u = y + num[o] * __builtin_amdgcn_rcpf(den[o] + EPSV);
                    float sg = 1.0f / (1.0f + __expf(-u));
                    out[o] = f2bf_bits(bf2f(x[o]) + u * sg);
                }
            }
        }
    }
}

// ---------------- edge kernel: z = x[i]W2 + x[j]W3 + eW4 + b; e += silu(z);
//                  num[i] += sigmoid(z)*h1[j]; den[i] += sigmoid(z)
// Wt points at W2; W3, W4 are contiguous after it (3 x 32 KB staged by one TDM op).

__global__ void edge_kernel(const unsigned short* __restrict__ x, unsigned short* e,
                            const int* __restrict__ idx_i, const int* __restrict__ idx_j,
                            const unsigned short* __restrict__ h1,
                            const unsigned short* __restrict__ Wt,
                            const float* __restrict__ b2, const float* __restrict__ b3,
                            const float* __restrict__ b4,
                            float* __restrict__ num, float* __restrict__ den, int E) {
    extern __shared__ char smem[];
    unsigned short* sW = (unsigned short*)smem;  // 3 x 128x128 bf16 (96 KB)
    stage_weights(Wt, smem, 3 * DIM * DIM);
    int wave = threadIdx.x >> 5, lane = threadIdx.x & 31;
    long tileBase = ((long)blockIdx.x * 8 + wave) * 16;
    if (tileBase >= E) return;                       // E is a multiple of 16
    int m = lane & 15, kh = lane >> 4;
    long erowA = tileBase + m;
    int gi = idx_i[erowA], gj = idx_j[erowA];
    const unsigned short* xi = x + (long)gi * DIM;
    const unsigned short* xj = x + (long)gj * DIM;
    const unsigned short* er = e + erowA * DIM;
    bf16x16 Ai[4], Aj[4], Ae[4];
    #pragma unroll
    for (int ks = 0; ks < 4; ++ks) {
        Ai[ks] = load_a_frag_bf(xi, ks * 32, kh);
        Aj[ks] = load_a_frag_bf(xj, ks * 32, kh);
        Ae[ks] = load_a_frag_bf(er, ks * 32, kh);
    }
    int n_ = lane & 15, rh = lane >> 4;
    int ii8[8], jj8[8];
    #pragma unroll
    for (int r = 0; r < 8; ++r) {
        long ed = tileBase + rh * 8 + r;
        ii8[r] = idx_i[ed];
        jj8[r] = idx_j[ed];
        __builtin_prefetch(h1 + (long)jj8[r] * DIM, 0, 0);  // global_prefetch_b8
    }
    for (int ct = 0; ct < 8; ++ct) {
        int col = ct * 16 + n_;
        float bv = b2[col] + b3[col] + b4[col];
        f32x8 acc;
        #pragma unroll
        for (int r = 0; r < 8; ++r) acc[r] = bv;
        #pragma unroll
        for (int ks = 0; ks < 4; ++ks) {
            acc = wmma_bf16(Ai[ks], load_b_frag(sW,         col, ks * 32, rh), acc);
            acc = wmma_bf16(Aj[ks], load_b_frag(sW + 16384, col, ks * 32, rh), acc);
            acc = wmma_bf16(Ae[ks], load_b_frag(sW + 32768, col, ks * 32, rh), acc);
        }
        #pragma unroll
        for (int r = 0; r < 8; ++r) {
            long ed = tileBase + rh * 8 + r;
            float z = acc[r];
            float sg = 1.0f / (1.0f + __expf(-z));
            long eo = ed * DIM + col;
            e[eo] = f2bf_bits(bf2f(e[eo]) + z * sg);              // e += silu(z)
            float hj = bf2f(h1[(long)jj8[r] * DIM + col]);
            atomicAdd(&num[(long)ii8[r] * DIM + col], sg * hj);
            atomicAdd(&den[(long)ii8[r] * DIM + col], sg);
        }
    }
}

// ---------------- readout ----------------

__global__ void pool_kernel(const unsigned short* __restrict__ h, const int* __restrict__ batch,
                            float* __restrict__ pooled, int N) {
    long idx = (long)blockIdx.x * blockDim.x + threadIdx.x;
    if (idx >= (long)N * DIM) return;
    int n = (int)(idx >> 7), d = (int)(idx & 127);
    atomicAdd(&pooled[batch[n] * DIM + d], bf2f(h[idx]));
}

__global__ void readout_kernel(const float* __restrict__ pooled, const float* __restrict__ headW,
                               const float* __restrict__ headb, const float* __restrict__ outW,
                               const float* __restrict__ outb, float* __restrict__ out) {
    extern __shared__ char smem[];
    unsigned short* sWt = (unsigned short*)smem;       // 32 KB: head_W^T bf16
    float* sH = (float*)(smem + 32768);                // 64 KB: silu(pooled@W+b)
    for (int i = threadIdx.x; i < DIM * DIM; i += 256) {
        int n = i >> 7, k = i & 127;
        sWt[i] = f2bf_bits(headW[k * DIM + n]);
    }
    __syncthreads();
    int wave = threadIdx.x >> 5, lane = threadIdx.x & 31;
    int m = lane & 15, kh = lane >> 4;
    const float* pr = pooled + (wave * 16 + m) * DIM;
    bf16x16 A[4];
    #pragma unroll
    for (int ks = 0; ks < 4; ++ks) A[ks] = load_a_frag_f32(pr, ks * 32, kh);
    int n_ = lane & 15, rh = lane >> 4;
    for (int ct = 0; ct < 8; ++ct) {
        int col = ct * 16 + n_;
        float bv = headb[col];
        f32x8 acc;
        #pragma unroll
        for (int r = 0; r < 8; ++r) acc[r] = bv;
        #pragma unroll
        for (int ks = 0; ks < 4; ++ks)
            acc = wmma_bf16(A[ks], load_b_frag(sWt, col, ks * 32, rh), acc);
        #pragma unroll
        for (int r = 0; r < 8; ++r) {
            int row = wave * 16 + rh * 8 + r;
            float y = acc[r];
            float sg = 1.0f / (1.0f + __expf(-y));
            sH[row * DIM + col] = y * sg;
        }
    }
    __syncthreads();
    for (int t = threadIdx.x; t < 128 * 3; t += 256) {
        int g = t / 3, o = t - g * 3;
        float s = outb[o];
        for (int k = 0; k < DIM; ++k) s += sH[g * DIM + k] * outW[k * 3 + o];
        out[t] = s;
    }
}

// ---------------- launch ----------------

extern "C" void kernel_launch(void* const* d_in, const int* in_sizes, int n_in,
                              void* d_out, int out_size, void* d_ws, size_t ws_size,
                              hipStream_t stream) {
    (void)in_sizes; (void)n_in; (void)out_size; (void)ws_size;
    const int*   x_atm   = (const int*)d_in[0];
    const float* x_bnd   = (const float*)d_in[1];
    const float* x_ang   = (const float*)d_in[2];
    const int*   eG      = (const int*)d_in[3];
    const int*   eA      = (const int*)d_in[4];
    const int*   batch   = (const int*)d_in[5];
    const float* embedW  = (const float*)d_in[6];
    const float* convA_W = (const float*)d_in[7];
    const float* convA_b = (const float*)d_in[8];
    const float* convG_W = (const float*)d_in[9];
    const float* convG_b = (const float*)d_in[10];
    const float* headW   = (const float*)d_in[11];
    const float* headb   = (const float*)d_in[12];
    const float* outW    = (const float*)d_in[13];
    const float* outb    = (const float*)d_in[14];

    const int N_ATM = 25000, N_EG = 400000, N_EA = 600000, NG = 128, L = 6;

    char* ws = (char*)d_ws;
    size_t off = 0;
    auto alloc = [&](size_t bytes) -> char* {
        char* p = ws + off;
        off += (bytes + 255) & ~(size_t)255;
        return p;
    };
    unsigned short* h_atm = (unsigned short*)alloc((size_t)N_ATM * DIM * 2);
    unsigned short* h_bnd = (unsigned short*)alloc((size_t)N_EG * DIM * 2);
    unsigned short* h_ang = (unsigned short*)alloc((size_t)N_EA * DIM * 2);
    unsigned short* h1    = (unsigned short*)alloc((size_t)N_EG * DIM * 2);
    float* numb  = (float*)alloc((size_t)N_EG * DIM * 4);
    float* denb  = (float*)alloc((size_t)N_EG * DIM * 4);
    unsigned short* WtA = (unsigned short*)alloc((size_t)L * 5 * DIM * DIM * 2);
    unsigned short* WtG = (unsigned short*)alloc((size_t)L * 5 * DIM * DIM * 2);
    float* pooled = (float*)alloc((size_t)NG * DIM * 4);

    {   // weights -> bf16 transposed
        int nm = L * 5;
        int blocks = (int)(((long)nm * DIM * DIM + 255) / 256);
        prep_weights_kernel<<<blocks, 256, 0, stream>>>(convA_W, WtA, nm);
        prep_weights_kernel<<<blocks, 256, 0, stream>>>(convG_W, WtG, nm);
    }
    embed_kernel<<<(N_ATM * DIM + 255) / 256, 256, 0, stream>>>(x_atm, embedW, h_atm, N_ATM);
    bessel_kernel<<<(int)(((long)N_EG * DIM + 255) / 256), 256, 0, stream>>>(x_bnd, h_bnd, N_EG);
    gaussian_kernel<<<(int)(((long)N_EA * DIM + 255) / 256), 256, 0, stream>>>(x_ang, h_ang, N_EA);

    const int* eA_i = eA;            const int* eA_j = eA + N_EA;
    const int* eG_i = eG;            const int* eG_j = eG + N_EG;
    const size_t MAT = (size_t)DIM * DIM;

    for (int it = 0; it < L; ++it) {
        // ---- conv A: x = h_bnd (N_EG nodes), e = h_ang, edges = N_EA
        const unsigned short* WA = WtA + (size_t)it * 5 * MAT;
        const float* bA = convA_b + (size_t)it * 5 * DIM;
        hipMemsetAsync(numb, 0, (size_t)N_EG * DIM * 4, stream);
        hipMemsetAsync(denb, 0, (size_t)N_EG * DIM * 4, stream);
        node_mm_kernel<<<(N_EG + 127) / 128, 256, 32768, stream>>>(
            h_bnd, WA + 1 * MAT, bA + 1 * DIM, h1, nullptr, nullptr, N_EG, 0);
        edge_kernel<<<(N_EA + 127) / 128, 256, 98304, stream>>>(
            h_bnd, h_ang, eA_i, eA_j, h1, WA + 2 * MAT,
            bA + 2 * DIM, bA + 3 * DIM, bA + 4 * DIM, numb, denb, N_EA);
        node_mm_kernel<<<(N_EG + 127) / 128, 256, 32768, stream>>>(
            h_bnd, WA, bA, h_bnd, numb, denb, N_EG, 1);

        // ---- conv G: x = h_atm (N_ATM nodes), e = h_bnd, edges = N_EG
        const unsigned short* WG = WtG + (size_t)it * 5 * MAT;
        const float* bG = convG_b + (size_t)it * 5 * DIM;
        hipMemsetAsync(numb, 0, (size_t)N_ATM * DIM * 4, stream);
        hipMemsetAsync(denb, 0, (size_t)N_ATM * DIM * 4, stream);
        node_mm_kernel<<<(N_ATM + 127) / 128, 256, 32768, stream>>>(
            h_atm, WG + 1 * MAT, bG + 1 * DIM, h1, nullptr, nullptr, N_ATM, 0);
        edge_kernel<<<(N_EG + 127) / 128, 256, 98304, stream>>>(
            h_atm, h_bnd, eG_i, eG_j, h1, WG + 2 * MAT,
            bG + 2 * DIM, bG + 3 * DIM, bG + 4 * DIM, numb, denb, N_EG);
        node_mm_kernel<<<(N_ATM + 127) / 128, 256, 32768, stream>>>(
            h_atm, WG, bG, h_atm, numb, denb, N_ATM, 1);
    }

    hipMemsetAsync(pooled, 0, (size_t)NG * DIM * 4, stream);
    pool_kernel<<<(N_ATM * DIM + 255) / 256, 256, 0, stream>>>(h_atm, batch, pooled, N_ATM);
    readout_kernel<<<1, 256, 98304, stream>>>(pooled, headW, headb, outW, outb, (float*)d_out);
}
